// MultiHeadSelfAttention_7791070675542
// MI455X (gfx1250) — matrix operand order repaired and checked by reading, not compiled, not run on hardware
//
#include <hip/hip_runtime.h>
#include <hip/hip_bf16.h>

#define D_MODEL 1024
#define SEQ     2048
#define BATCH   4
#define NHEAD   16
#define HDIM    64
#define BS      (BATCH*SEQ)

typedef __attribute__((ext_vector_type(16))) __bf16 v16bf;
typedef __attribute__((ext_vector_type(8)))  float  v8f;
typedef __attribute__((ext_vector_type(4))) unsigned int v4u;
typedef __attribute__((ext_vector_type(8))) int v8i;
typedef __attribute__((ext_vector_type(4))) int v4i;
typedef unsigned short bf16_t;

union FragBF { v16bf v; uint4 u[2]; };
union FragF  { v8f   v; float f[8]; };

__device__ __forceinline__ bf16_t f2bf(float f) {
  union { float f; unsigned u; } c; c.f = f;
  unsigned u = c.u + 0x7FFFu + ((c.u >> 16) & 1u);   // round-to-nearest-even
  return (bf16_t)(u >> 16);
}

// A-matrix 16x32 bf16 fragment (ISA 7.12.2): lane<16 holds K=k0..k0+7 (v0..3)
// and K=k0+16..k0+23 (v4..7); lane>=16 holds K=k0+8..15 and k0+24..31.
__device__ __forceinline__ FragBF load_a_frag(const bf16_t* A, int lda,
                                              int row0, int k0, int lane) {
  int mm = lane & 15, hi = lane >> 4;
  const bf16_t* p = A + (size_t)(row0 + mm) * lda + k0 + hi * 8;
  FragBF r;
  r.u[0] = *(const uint4*)p;
  r.u[1] = *(const uint4*)(p + 16);
  return r;
}

// B-matrix 32x16 bf16 fragment, B[k][n] read from R where R[n][k] = B[k][n]
// (row-major, stride ldr). lanes 0-15 hold K=k0..k0+7 and k0+16..k0+23 for col n=lane;
// lanes 16-31 hold K=k0+8..k0+15 and k0+24..k0+31.
__device__ __forceinline__ FragBF load_bt_frag(const bf16_t* R, int ldr,
                                               int n0, int k0, int lane) {
  int nn = lane & 15, hi = lane >> 4;
  const bf16_t* p = R + (size_t)(n0 + nn) * ldr + k0 + hi * 8;
  FragBF r;
  r.u[0] = *(const uint4*)p;
  r.u[1] = *(const uint4*)(p + 16);
  return r;
}

__device__ __forceinline__ v8f wmma_bf16(const FragBF& a, const FragBF& b, v8f c) {
  return __builtin_amdgcn_wmma_f32_16x16x32_bf16(false, a.v, false, b.v,
                                                 (short)0, c, false, false);
}

// ---- Tensor Data Mover: DMA a 128-row x 32-col bf16 tile (row-major,
// leading dim ld_elems) from global memory into LDS (rows packed, 64B/row).
// D# layout per CDNA5 ISA ch.8: group0 = count/lds_addr/global_addr/type,
// group1 = data_size + tensor dims + tile dims + strides. 2D => groups 2/3 zero.
__device__ __forceinline__ void tdm_load_tile128x32(bf16_t* lds_dst,
                                                    const bf16_t* gsrc,
                                                    unsigned ld_elems) {
  unsigned lds_addr = (unsigned)(uintptr_t)lds_dst;   // low 32b of flat = LDS byte offset
  unsigned long long ga = (unsigned long long)(uintptr_t)gsrc;
  v4u g0;
  g0[0] = 1u;                                           // count=1 (valid user D#)
  g0[1] = lds_addr;                                     // lds_addr [63:32]
  g0[2] = (unsigned)ga;                                 // global_addr[31:0]
  g0[3] = (unsigned)((ga >> 32) & 0x1FFFFFFull) | (2u << 30); // addr[56:32] | type=2
  v8i g1;
  g1[0] = (int)(1u << 16);                              // wg_mask=0, data_size=1 (2B)
  g1[1] = (int)((ld_elems & 0xFFFFu) << 16);            // tensor_dim0[15:0]
  g1[2] = (int)((ld_elems >> 16) | (128u << 16));       // tensor_dim0[31:16]|tensor_dim1=128
  g1[3] = (int)(32u << 16);                             // tensor_dim1 hi=0 | tile_dim0=32
  g1[4] = (int)128u;                                    // tile_dim1=128, tile_dim2=0
  g1[5] = (int)ld_elems;                                // tensor_dim0_stride[31:0]
  g1[6] = 0;                                            // stride hi | dim1_stride lo
  g1[7] = 0;
  v4i z4 = {0, 0, 0, 0};
  v8i z8 = {0, 0, 0, 0, 0, 0, 0, 0};
  // 6-arg form (clang-23 / therock-10.0 headers): groups 0..3 + extra group + cpol
  __builtin_amdgcn_tensor_load_to_lds(g0, g1, z4, z4, z8, 0);
}

// ---------------- conversion kernels ----------------
__global__ void k_cvt(const float* __restrict__ in, bf16_t* __restrict__ out, int n) {
  for (int i = blockIdx.x * blockDim.x + threadIdx.x; i < n; i += gridDim.x * blockDim.x)
    out[i] = f2bf(in[i]);
}

// out[n*1024 + k] = bf16(in[k*1024 + n])  (store W transposed)
__global__ void k_transpose_cvt(const float* __restrict__ in, bf16_t* __restrict__ out) {
  int idx = blockIdx.x * blockDim.x + threadIdx.x;
  if (idx < D_MODEL * D_MODEL) {
    int n = idx >> 10, k = idx & 1023;
    out[idx] = f2bf(in[k * D_MODEL + n]);
  }
}

// ---------------- fused QKV projection + bias + RoPE ----------------
// grid: (BS/128, D_MODEL/128, 3); block 256 (8 waves). z: 0=Q,1=K,2=V
// TDM double-buffered LDS pipeline: wave 0 DMAs the next A/B 128x32 tiles
// while all 8 waves run WMMA on the current tiles.
__global__ void k_qkv_gemm(const bf16_t* __restrict__ xb,
                           const bf16_t* __restrict__ WqT, const bf16_t* __restrict__ WkT,
                           const bf16_t* __restrict__ WvT,
                           const float* __restrict__ bq, const float* __restrict__ bk,
                           const float* __restrict__ bv,
                           bf16_t* __restrict__ qb, bf16_t* __restrict__ kb,
                           bf16_t* __restrict__ vT) {
  __shared__ __align__(16) bf16_t lA[2][128 * 32];
  __shared__ __align__(16) bf16_t lB[2][128 * 32];
  const int z = blockIdx.z;
  const bf16_t* WT  = (z == 0) ? WqT : (z == 1) ? WkT : WvT;
  const float* bias = (z == 0) ? bq  : (z == 1) ? bk  : bv;
  const int lane = threadIdx.x & 31;
  const int wave = threadIdx.x >> 5;
  const int row0 = blockIdx.x * 128;
  const int col0 = blockIdx.y * 128;
  const int wm = (wave >> 2) * 64;   // local m base: 4 m-tiles of 16
  const int wn = (wave & 3) * 32;    // local n base: 2 n-tiles of 16

  if (wave == 0) {
    tdm_load_tile128x32(lA[0], xb + (size_t)row0 * D_MODEL, D_MODEL);
    tdm_load_tile128x32(lB[0], WT + (size_t)col0 * D_MODEL, D_MODEL);
  }

  FragF acc[4][2] = {};
  for (int k0 = 0; k0 < D_MODEL; k0 += 32) {
    const int buf = (k0 >> 5) & 1;
    if (wave == 0) {
      if (k0 + 32 < D_MODEL) {
        tdm_load_tile128x32(lA[buf ^ 1], xb + (size_t)row0 * D_MODEL + k0 + 32, D_MODEL);
        tdm_load_tile128x32(lB[buf ^ 1], WT + (size_t)col0 * D_MODEL + k0 + 32, D_MODEL);
        __builtin_amdgcn_s_wait_tensorcnt(2);  // current buf's 2 DMAs retired
      } else {
        __builtin_amdgcn_s_wait_tensorcnt(0);
      }
    }
    __syncthreads();
    FragBF a[4], b[2];
#pragma unroll
    for (int i = 0; i < 4; i++) a[i] = load_a_frag(lA[buf], 32, wm + i * 16, 0, lane);
#pragma unroll
    for (int j = 0; j < 2; j++) b[j] = load_bt_frag(lB[buf], 32, wn + j * 16, 0, lane);
#pragma unroll
    for (int i = 0; i < 4; i++)
#pragma unroll
      for (int j = 0; j < 2; j++)
        acc[i][j].v = wmma_bf16(a[i], b[j], acc[i][j].v);
    __syncthreads();  // all reads of buf done before it is re-filled
  }

  const int hi = lane >> 4;
#pragma unroll
  for (int i = 0; i < 4; i++)
#pragma unroll
    for (int j = 0; j < 2; j++)
#pragma unroll
      for (int v = 0; v < 8; v++) {
        int grow = row0 + wm + i * 16 + v + 8 * hi;   // row in [0, BS)
        int gcol = col0 + wn + j * 16 + (lane & 15);  // col in [0, D_MODEL)
        float val = acc[i][j].f[v] + bias[gcol];
        int bb = grow >> 11, s = grow & (SEQ - 1);
        int h  = gcol >> 6,  dd = gcol & 63;
        if (z < 2) {
          // RoPE: paired column lives in adjacent lane (same m, n^1)
          float partner = __shfl_xor(val, 1, 32);
          float rot = (gcol & 1) ? partner : -partner;
          float ang = (float)s * __powf(10000.f, -(float)(dd & ~1) * (1.f / 64.f));
          float sn, cs; __sincosf(ang, &sn, &cs);
          val = val * cs + rot * sn;
          bf16_t* dst = (z == 0) ? qb : kb;
          dst[((size_t)(bb * NHEAD + h) * SEQ + s) * HDIM + dd] = f2bf(val);
        } else {
          // V stored transposed: [B,H,D,S]
          vT[((size_t)(bb * NHEAD + h) * HDIM + dd) * SEQ + s] = f2bf(val);
        }
      }
}

// ---------------- flash attention (causal, online softmax) ----------------
// grid: (B*H, SEQ/128); block 256 (8 waves); each wave owns a 16-row Q tile.
__global__ void k_flash_attn(const bf16_t* __restrict__ qb, const bf16_t* __restrict__ kb,
                             const bf16_t* __restrict__ vT, bf16_t* __restrict__ yb) {
  const int bh   = blockIdx.x;
  const int lane = threadIdx.x & 31;
  const int wave = threadIdx.x >> 5;
  const int q0   = (blockIdx.y * 8 + wave) * 16;
  const bf16_t* Q = qb + (size_t)bh * SEQ * HDIM;
  const bf16_t* K = kb + (size_t)bh * SEQ * HDIM;
  const bf16_t* V = vT + (size_t)bh * HDIM * SEQ;
  const int mm = lane & 15, hi = lane >> 4;

  __shared__ __align__(16) bf16_t pshare[8][16 * 32];  // per-wave P staging
  bf16_t* P = &pshare[wave][0];

  FragBF aq0 = load_a_frag(Q, HDIM, q0, 0, lane);
  FragBF aq1 = load_a_frag(Q, HDIM, q0, 32, lane);
  FragF o[4] = {};
  float mi[8], li[8];
#pragma unroll
  for (int v = 0; v < 8; v++) { mi[v] = -1e30f; li[v] = 0.f; }
  const FragF zf = {};
  const float scale = 0.125f;  // 1/sqrt(64)

  for (int kv = 0; kv < q0 + 16; kv += 32) {
    FragF st[2];
#pragma unroll
    for (int t = 0; t < 2; t++) {
      FragBF bk0 = load_bt_frag(K, HDIM, kv + 16 * t, 0, lane);
      FragBF bk1 = load_bt_frag(K, HDIM, kv + 16 * t, 32, lane);
      st[t].v = wmma_bf16(aq0, bk0, zf.v);
      st[t].v = wmma_bf16(aq1, bk1, st[t].v);
    }
    float nm[8], ps[2][8];
#pragma unroll
    for (int v = 0; v < 8; v++) nm[v] = mi[v];
#pragma unroll
    for (int t = 0; t < 2; t++)
#pragma unroll
      for (int v = 0; v < 8; v++) {
        float x = st[t].f[v] * scale;
        int col = kv + 16 * t + mm;
        int row = q0 + v + 8 * hi;
        if (col > row) x = -1e30f;        // causal mask
        ps[t][v] = x;
        nm[v] = fmaxf(nm[v], x);
      }
#pragma unroll
    for (int mask = 1; mask < 16; mask <<= 1)
#pragma unroll
      for (int v = 0; v < 8; v++)
        nm[v] = fmaxf(nm[v], __shfl_xor(nm[v], mask, 32));

    float rs[8];
#pragma unroll
    for (int v = 0; v < 8; v++) {
      float a = __expf(mi[v] - nm[v]);
      li[v] *= a;
#pragma unroll
      for (int dt = 0; dt < 4; dt++) o[dt].f[v] *= a;
      mi[v] = nm[v];
      float p0 = __expf(ps[0][v] - nm[v]);
      float p1 = __expf(ps[1][v] - nm[v]);
      rs[v] = p0 + p1;
      P[(v + 8 * hi) * 32 + mm]      = f2bf(p0);   // row-major 16x32 stage
      P[(v + 8 * hi) * 32 + 16 + mm] = f2bf(p1);
    }
#pragma unroll
    for (int mask = 1; mask < 16; mask <<= 1)
#pragma unroll
      for (int v = 0; v < 8; v++)
        rs[v] += __shfl_xor(rs[v], mask, 32);
#pragma unroll
    for (int v = 0; v < 8; v++) li[v] += rs[v];

    // LDS is in-order per wave; drain stores before cross-lane re-read
    asm volatile("s_wait_dscnt 0" ::: "memory");

    FragBF pa;  // re-read P in A-fragment layout
    {
      const bf16_t* pr = P + mm * 32;
      pa.u[0] = *(const uint4*)(pr + hi * 8);
      pa.u[1] = *(const uint4*)(pr + 16 + hi * 8);
    }
#pragma unroll
    for (int dt = 0; dt < 4; dt++) {
      FragBF bv = load_bt_frag(V, SEQ, dt * 16, kv, lane);  // V^T rows = head dims
      o[dt].v = wmma_bf16(pa, bv, o[dt].v);
    }
  }

  const int bb = bh >> 4, h = bh & 15;
#pragma unroll
  for (int dt = 0; dt < 4; dt++)
#pragma unroll
    for (int v = 0; v < 8; v++) {
      float val = o[dt].f[v] / li[v];
      int s = q0 + v + 8 * hi;
      int gcol = h * HDIM + dt * 16 + mm;
      yb[(size_t)(bb * SEQ + s) * D_MODEL + gcol] = f2bf(val);
    }
}

// ---------------- output projection (TDM-fed, same pipeline) ----------------
__global__ void k_out_gemm(const bf16_t* __restrict__ yb, const bf16_t* __restrict__ WoT,
                           const float* __restrict__ bo, float* __restrict__ out) {
  __shared__ __align__(16) bf16_t lA[2][128 * 32];
  __shared__ __align__(16) bf16_t lB[2][128 * 32];
  const int lane = threadIdx.x & 31;
  const int wave = threadIdx.x >> 5;
  const int row0 = blockIdx.x * 128;
  const int col0 = blockIdx.y * 128;
  const int wm = (wave >> 2) * 64;
  const int wn = (wave & 3) * 32;

  if (wave == 0) {
    tdm_load_tile128x32(lA[0], yb + (size_t)row0 * D_MODEL, D_MODEL);
    tdm_load_tile128x32(lB[0], WoT + (size_t)col0 * D_MODEL, D_MODEL);
  }

  FragF acc[4][2] = {};
  for (int k0 = 0; k0 < D_MODEL; k0 += 32) {
    const int buf = (k0 >> 5) & 1;
    if (wave == 0) {
      if (k0 + 32 < D_MODEL) {
        tdm_load_tile128x32(lA[buf ^ 1], yb + (size_t)row0 * D_MODEL + k0 + 32, D_MODEL);
        tdm_load_tile128x32(lB[buf ^ 1], WoT + (size_t)col0 * D_MODEL + k0 + 32, D_MODEL);
        __builtin_amdgcn_s_wait_tensorcnt(2);
      } else {
        __builtin_amdgcn_s_wait_tensorcnt(0);
      }
    }
    __syncthreads();
    FragBF a[4], b[2];
#pragma unroll
    for (int i = 0; i < 4; i++) a[i] = load_a_frag(lA[buf], 32, wm + i * 16, 0, lane);
#pragma unroll
    for (int j = 0; j < 2; j++) b[j] = load_bt_frag(lB[buf], 32, wn + j * 16, 0, lane);
#pragma unroll
    for (int i = 0; i < 4; i++)
#pragma unroll
      for (int j = 0; j < 2; j++)
        acc[i][j].v = wmma_bf16(a[i], b[j], acc[i][j].v);
    __syncthreads();
  }

  const int hi = lane >> 4;
#pragma unroll
  for (int i = 0; i < 4; i++)
#pragma unroll
    for (int j = 0; j < 2; j++)
#pragma unroll
      for (int v = 0; v < 8; v++) {
        int grow = row0 + wm + i * 16 + v + 8 * hi;
        int gcol = col0 + wn + j * 16 + (lane & 15);
        out[(size_t)grow * D_MODEL + gcol] = acc[i][j].f[v] + bo[gcol];
      }
}

extern "C" void kernel_launch(void* const* d_in, const int* in_sizes, int n_in,
                              void* d_out, int out_size, void* d_ws, size_t ws_size,
                              hipStream_t stream) {
  const float* x  = (const float*)d_in[0];
  const float* Wq = (const float*)d_in[1];
  const float* bq = (const float*)d_in[2];
  const float* Wk = (const float*)d_in[3];
  const float* bk = (const float*)d_in[4];
  const float* Wv = (const float*)d_in[5];
  const float* bv = (const float*)d_in[6];
  const float* Wo = (const float*)d_in[7];
  const float* bo = (const float*)d_in[8];
  float* out = (float*)d_out;

  char* ws = (char*)d_ws;
  const size_t MB = 1024ull * 1024ull;
  bf16_t* xb  = (bf16_t*)(ws);             // 16 MB
  bf16_t* WqT = (bf16_t*)(ws + 16 * MB);   //  2 MB each
  bf16_t* WkT = (bf16_t*)(ws + 18 * MB);
  bf16_t* WvT = (bf16_t*)(ws + 20 * MB);
  bf16_t* WoT = (bf16_t*)(ws + 22 * MB);
  bf16_t* qb  = (bf16_t*)(ws + 24 * MB);   // 16 MB  [B,H,S,D]
  bf16_t* kb  = (bf16_t*)(ws + 40 * MB);   // 16 MB  [B,H,S,D]
  bf16_t* vT  = (bf16_t*)(ws + 56 * MB);   // 16 MB  [B,H,D,S]
  bf16_t* yb  = (bf16_t*)(ws + 72 * MB);   // 16 MB  [B*S, D_MODEL]

  k_cvt<<<4096, 256, 0, stream>>>(x, xb, BS * D_MODEL);
  k_transpose_cvt<<<4096, 256, 0, stream>>>(Wq, WqT);
  k_transpose_cvt<<<4096, 256, 0, stream>>>(Wk, WkT);
  k_transpose_cvt<<<4096, 256, 0, stream>>>(Wv, WvT);
  k_transpose_cvt<<<4096, 256, 0, stream>>>(Wo, WoT);
  k_qkv_gemm<<<dim3(BS / 128, D_MODEL / 128, 3), 256, 0, stream>>>(
      xb, WqT, WkT, WvT, bq, bk, bv, qb, kb, vT);
  k_flash_attn<<<dim3(BATCH * NHEAD, SEQ / 128), 256, 0, stream>>>(qb, kb, vT, yb);
  k_out_gemm<<<dim3(BS / 128, D_MODEL / 128), 256, 0, stream>>>(yb, WoT, bo, out);
}
